// LlamaSelfAttentionBlock_21655225106591
// MI455X (gfx1250) — matrix operand order repaired and checked by reading, not compiled, not run on hardware
//
#include <hip/hip_runtime.h>
#include <cstdint>
#include <cstddef>

// ---------------------------------------------------------------------------
// CDNA5 (gfx1250) bf16 WMMA implementation of the Llama self-attention block.
// All heavy GEMMs + attention use v_wmma_f32_16x16x32_bf16 (wave32).
// GEMM tiles are staged with gfx1250 async-to-LDS loads (ASYNCcnt path).
// ---------------------------------------------------------------------------

typedef __bf16 bf16_t;
typedef __attribute__((ext_vector_type(16))) __bf16 v16bf;
typedef __attribute__((ext_vector_type(8)))  float  v8f;
typedef int v4i_vs __attribute__((vector_size(16)));  // matches builtin's V4i

static __device__ __forceinline__ v8f wmma_bf16(v16bf a, v16bf b, v8f c) {
  // 8 args: (neg_a, A, neg_b, B, c_mod, C, reuse_a, reuse_b)
  return __builtin_amdgcn_wmma_f32_16x16x32_bf16(false, a, false, b, (short)0, c,
                                                 false, false);
}

union BFrag {
  v16bf  v;
  float4 f4[2];   // 2 x 16B chunks (8 bf16 each)
};

// --- async global->LDS staging (gfx1250 ASYNCcnt path), with sync fallback --
#if __has_builtin(__builtin_amdgcn_global_load_async_to_lds_b128) && \
    __has_builtin(__builtin_amdgcn_s_wait_asynccnt)
#define USE_ASYNC_LDS 1
#define STAGE16(ldsdst, gsrc)                                                   \
  __builtin_amdgcn_global_load_async_to_lds_b128(                               \
      (__attribute__((address_space(1))) v4i_vs*)(uintptr_t)(gsrc),             \
      (__attribute__((address_space(3))) v4i_vs*)(uint32_t)(uintptr_t)(ldsdst), \
      0, 0)
#define WAIT_ASYNC() __builtin_amdgcn_s_wait_asynccnt(0)
#else
#define USE_ASYNC_LDS 0
#define STAGE16(ldsdst, gsrc) (*(float4*)(ldsdst) = *(const float4*)(gsrc))
#define WAIT_ASYNC() ((void)0)
#endif

// ---------------------------------------------------------------------------
// fp32 -> bf16 elementwise conversion (used once per call for weights)
// ---------------------------------------------------------------------------
__global__ __launch_bounds__(256)
void f32_to_bf16_kernel(const float* __restrict__ in, bf16_t* __restrict__ out,
                        size_t n) {
  size_t i = (size_t)blockIdx.x * blockDim.x + threadIdx.x;
  const size_t stride = (size_t)gridDim.x * blockDim.x;
  for (; i < n; i += stride) out[i] = (bf16_t)in[i];
}

// ---------------------------------------------------------------------------
// RoPE diagonal table: tab[l*HD + d] = cos(l * 10000^(-2*(d/2)/HD))
// ---------------------------------------------------------------------------
__global__ __launch_bounds__(256)
void rope_table_kernel(float* __restrict__ tab, int L, int HD) {
  const int i = blockIdx.x * blockDim.x + threadIdx.x;
  if (i < L * HD) {
    const int l = i / HD, d = i % HD;
    const int pair = d >> 1;
    const float theta =
        expf((float)pair * (-2.0f / (float)HD) * 9.2103403719761836f);
    tab[i] = cosf((float)l * theta);
  }
}

// ---------------------------------------------------------------------------
// RMSNorm (one 256-thread block per row) -> bf16 output
// ---------------------------------------------------------------------------
__global__ __launch_bounds__(256)
void rmsnorm_bf16_kernel(const float* __restrict__ x, const float* __restrict__ g,
                         bf16_t* __restrict__ out, int E) {
  const int row = blockIdx.x;
  const float* xr = x + (size_t)row * E;
  float ss = 0.f;
  for (int i = threadIdx.x; i < E; i += 256) { float v = xr[i]; ss += v * v; }
  __shared__ float red[256];
  red[threadIdx.x] = ss;
  __syncthreads();
  for (int s = 128; s > 0; s >>= 1) {
    if (threadIdx.x < s) red[threadIdx.x] += red[threadIdx.x + s];
    __syncthreads();
  }
  const float rr = rsqrtf(red[0] / (float)E + 1e-6f);
  bf16_t* orow = out + (size_t)row * E;
  for (int i = threadIdx.x; i < E; i += 256) orow[i] = (bf16_t)(xr[i] * rr * g[i]);
}

// ---------------------------------------------------------------------------
// SwiGLU: act = gate * sigmoid(beta*gate) * lin  (in place over gate)
// ---------------------------------------------------------------------------
__global__ __launch_bounds__(256)
void swiglu_kernel(bf16_t* __restrict__ gate, const bf16_t* __restrict__ lin,
                   const float* __restrict__ beta, size_t n) {
  const float b0 = beta[0];
  size_t i = (size_t)blockIdx.x * blockDim.x + threadIdx.x;
  const size_t stride = (size_t)gridDim.x * blockDim.x;
  for (; i < n; i += stride) {
    float gv = (float)gate[i];
    float s  = 1.f / (1.f + __expf(-b0 * gv));
    gate[i]  = (bf16_t)(gv * s * (float)lin[i]);
  }
}

// ---------------------------------------------------------------------------
// Generic bf16 GEMM:  C[M,N] = A[M,K] @ B[K,N]  (row-major, fp32 accumulate)
// Block tile 128x128, BK=32, 8 waves; each wave: 32x64 via 8 WMMA per K-step.
// Double-buffered LDS; tiles staged via async-to-LDS when available.
// Epilogue modes fuse bias / RoPE-table / layout scatter / residual.
// ---------------------------------------------------------------------------
enum GemmEp {
  EP_F32_BIAS_RES = 0,  // outF = acc + bias[n] + res[m*N+n]   (fp32)
  EP_BF16_BIAS    = 1,  // outB = bf16(acc + bias[n])          (row-major)
  EP_ROPE_Q       = 2,  // q[b,h,l,d]  = bf16(acc * rope[l,d])
  EP_ROPE_KT      = 3,  // kT[b,h,d,l] = bf16(acc * rope[l,d])
  EP_VHEAD        = 4   // v[b,h,l,d]  = bf16(acc)
};

template <int MODE>
__global__ __launch_bounds__(256)
void gemm_bf16_kernel(int M, int N, int K,
                      const bf16_t* __restrict__ A,
                      const bf16_t* __restrict__ Bw,
                      const float* __restrict__ bias,
                      const float* __restrict__ res,
                      const float* __restrict__ rope,
                      float* __restrict__ outF,
                      bf16_t* __restrict__ outB,
                      int L, int Hn, int HD) {
  __shared__ alignas(16) bf16_t Alds[2][128 * 32];
  __shared__ alignas(16) bf16_t Blds[2][32 * 128];

  const int tid  = threadIdx.x;
  const int lane = tid & 31;
  const int w    = tid >> 5;
  const int hl   = lane >> 4;       // half-wave select (K/M split per ISA layout)
  const int ln   = lane & 15;
  const int wmo  = (w & 3) * 32;    // wave row offset within block tile
  const int wno  = (w >> 2) * 64;   // wave col offset within block tile
  const int tm   = blockIdx.y * 128;
  const int tn   = blockIdx.x * 128;

  // Per-thread staging chunk coordinates (2 x 16B for A tile, 2 x 16B for B)
  const int ar0 = tid >> 2,         ac0 = (tid & 3) * 8;
  const int ar1 = (tid + 256) >> 2, ac1 = ((tid + 256) & 3) * 8;
  const int br0 = tid >> 4,         bc0 = (tid & 15) * 8;
  const int br1 = (tid + 256) >> 4, bc1 = ((tid + 256) & 15) * 8;

  auto stage = [&](int buf, int k0) {
    STAGE16(&Alds[buf][ar0 * 32 + ac0], A + (size_t)(tm + ar0) * K + k0 + ac0);
    STAGE16(&Alds[buf][ar1 * 32 + ac1], A + (size_t)(tm + ar1) * K + k0 + ac1);
    STAGE16(&Blds[buf][br0 * 128 + bc0], Bw + (size_t)(k0 + br0) * N + tn + bc0);
    STAGE16(&Blds[buf][br1 * 128 + bc1], Bw + (size_t)(k0 + br1) * N + tn + bc1);
  };

  v8f acc[2][4] = {};

  stage(0, 0);
  int buf = 0;
  for (int k0 = 0; k0 < K; k0 += 32) {
    WAIT_ASYNC();          // this wave's staged tile is in LDS
    __syncthreads();       // ... and everyone else's too
    if (k0 + 32 < K) stage(buf ^ 1, k0 + 32);  // overlap next tile with compute

    // A fragments: lane ln holds row, K elements {hl*8+0..7, hl*8+16..23}
    BFrag a[2];
#pragma unroll
    for (int mf = 0; mf < 2; ++mf) {
      const bf16_t* p = &Alds[buf][(wmo + mf * 16 + ln) * 32 + hl * 8];
      a[mf].f4[0] = *(const float4*)p;
      a[mf].f4[1] = *(const float4*)(p + 16);
    }
    // B fragments: lane holds N-col = ln, K elements {hl*8+0..7, hl*8+16..23}
#pragma unroll
    for (int nf = 0; nf < 4; ++nf) {
      BFrag b;
#pragma unroll
      for (int s = 0; s < 16; ++s)
        b.v[s] = Blds[buf][((s < 8) ? hl * 8 + s : 16 + hl * 8 + s - 8) * 128 + wno + nf * 16 + ln];
#pragma unroll
      for (int mf = 0; mf < 2; ++mf)
        acc[mf][nf] = wmma_bf16(a[mf].v, b.v, acc[mf][nf]);
    }
    buf ^= 1;
  }

  // Epilogue: C layout -> lane col = ln, rows = hl*8 + r
#pragma unroll
  for (int mf = 0; mf < 2; ++mf) {
#pragma unroll
    for (int nf = 0; nf < 4; ++nf) {
      const int n = tn + wno + nf * 16 + ln;
#pragma unroll
      for (int r = 0; r < 8; ++r) {
        const int m  = tm + wmo + mf * 16 + hl * 8 + r;
        const float vv = acc[mf][nf][r];
        if constexpr (MODE == EP_F32_BIAS_RES) {
          const size_t idx = (size_t)m * N + n;
          outF[idx] = vv + bias[n] + res[idx];
        } else if constexpr (MODE == EP_BF16_BIAS) {
          outB[(size_t)m * N + n] = (bf16_t)(vv + bias[n]);
        } else {
          const int b = m / L, l = m % L;
          const int h = n / HD, d = n % HD;
          if constexpr (MODE == EP_VHEAD) {
            outB[(((size_t)(b * Hn + h)) * L + l) * HD + d] = (bf16_t)vv;
          } else {
            const float sc = rope[(size_t)l * HD + d];
            if constexpr (MODE == EP_ROPE_Q)
              outB[(((size_t)(b * Hn + h)) * L + l) * HD + d] = (bf16_t)(vv * sc);
            else  // EP_ROPE_KT
              outB[(((size_t)(b * Hn + h)) * HD + d) * L + l] = (bf16_t)(vv * sc);
          }
        }
      }
    }
  }
}

// ---------------------------------------------------------------------------
// Causal attention (flash-style): 16 query rows per block, 512-key chunks.
// Online softmax with running max/sum; S=Q@K^T and O=P@V via bf16 WMMA.
// x1 = O + x (residual fused). ~51KB static LDS.
// ---------------------------------------------------------------------------
__global__ __launch_bounds__(256)
void attention_kernel(const bf16_t* __restrict__ q, const bf16_t* __restrict__ kT,
                      const bf16_t* __restrict__ v, const float* __restrict__ xres,
                      float* __restrict__ x1, int L, int Hn, int HD, int E) {
  constexpr int SSTR = 516;   // fp32 score row stride
  constexpr int PSTR = 520;   // bf16 P row stride (byte stride 1040 = 65*16)
  constexpr int KC   = 512;   // keys per chunk
  __shared__ alignas(16) float  S[16 * SSTR];
  __shared__ alignas(16) bf16_t P[16 * PSTR];
  __shared__ float red[256];
  __shared__ float rm[16], rs[16], ff[16];

  const int tid  = threadIdx.x;
  const int lane = tid & 31;
  const int w    = tid >> 5;
  const int hl   = lane >> 4;
  const int ln   = lane & 15;
  const int qt   = blockIdx.x;
  const int bh   = blockIdx.y;
  const int l0   = qt * 16;
  const int kmax = l0 + 16;               // causal limit (exclusive)
  const float scale = 0.08838834764831845f;  // 1/sqrt(128)

  // Preload Q fragments for the 16 query rows (hd=128 -> 4 K-steps)
  BFrag qa[4];
  {
    const bf16_t* qrow = q + ((size_t)bh * L + l0 + ln) * HD + hl * 8;
#pragma unroll
    for (int kk = 0; kk < 4; ++kk) {
      qa[kk].f4[0] = *(const float4*)(qrow + kk * 32);
      qa[kk].f4[1] = *(const float4*)(qrow + kk * 32 + 16);
    }
  }
  if (tid < 16) { rm[tid] = -INFINITY; rs[tid] = 0.f; }
  __syncthreads();

  v8f o = {};
  const int nchunks = (kmax + KC - 1) / KC;
  for (int c = 0; c < nchunks; ++c) {
    const int kbase = c * KC;
    const int kend  = min(kmax, kbase + KC);
    const int nkt32 = (kend - kbase + 31) >> 5;
    const int nkt16 = nkt32 * 2;
    const int ncol  = nkt32 * 32;

    // ---- Phase 1: S = (Q K^T) * scale, causal-masked, into LDS ----
    for (int kt = w; kt < nkt16; kt += 8) {
      const int n0 = kt * 16;
      if (kt + 8 < nkt16)  // prefetch this wave's next K^T tile
        __builtin_prefetch(kT + ((size_t)bh * HD + lane) * L + kbase + (kt + 8) * 16,
                           0, 0);
      v8f s = {};
#pragma unroll
      for (int kk = 0; kk < 4; ++kk) {
        BFrag kb;
#pragma unroll
        for (int s = 0; s < 16; ++s)
          kb.v[s] = kT[((size_t)bh * HD + kk * 32 + ((s < 8) ? hl * 8 + s : 16 + hl * 8 + s - 8)) * L + kbase + n0 + ln];
        s = wmma_bf16(qa[kk].v, kb.v, s);
      }
      const int kg = kbase + n0 + ln;  // global key index for this lane's column
#pragma unroll
      for (int r = 0; r < 8; ++r) {
        const int m = hl * 8 + r;      // local query row
        S[m * SSTR + n0 + ln] = (kg <= l0 + m) ? s[r] * scale : -INFINITY;
      }
    }
    __syncthreads();

    // ---- Phase 2: online softmax update for this chunk ----
    const int row = tid >> 4, sub = tid & 15;
    float pmax = -INFINITY;
    for (int n = sub; n < ncol; n += 16) pmax = fmaxf(pmax, S[row * SSTR + n]);
    red[row * 16 + sub] = pmax;
    __syncthreads();
    if (sub == 0) {
      float mc = red[row * 16];
      for (int j = 1; j < 16; ++j) mc = fmaxf(mc, red[row * 16 + j]);
      const float mold = rm[row];
      const float mnew = fmaxf(mold, mc);      // mc finite -> mnew finite
      ff[row] = __expf(mold - mnew);           // 0 on first chunk (mold=-inf)
      rm[row] = mnew;
    }
    __syncthreads();
    const float mnew = rm[row];
    float psum = 0.f;
    for (int n = sub; n < ncol; n += 16) {
      const float e = __expf(S[row * SSTR + n] - mnew);
      P[row * PSTR + n] = (bf16_t)e;
      psum += e;
    }
    red[row * 16 + sub] = psum;
    __syncthreads();
    if (sub == 0) {
      float sm = 0.f;
      for (int j = 0; j < 16; ++j) sm += red[row * 16 + j];
      rs[row] = rs[row] * ff[row] + sm;
    }
    __syncthreads();

    // ---- Phase 3: O = O*f + P @ V (wave w owns head dims [w*16, w*16+16)) ----
#pragma unroll
    for (int r = 0; r < 8; ++r) o[r] *= ff[hl * 8 + r];
    for (int ks = 0; ks < nkt32; ++ks) {
      if (ks + 1 < nkt32)  // prefetch next V tile
        __builtin_prefetch(
            v + ((size_t)bh * L + kbase + (ks + 1) * 32 + lane) * HD + w * 16, 0, 0);
      BFrag pa, vb;
      const bf16_t* pp = P + ln * PSTR + ks * 32 + hl * 8;
      pa.f4[0] = *(const float4*)pp;
      pa.f4[1] = *(const float4*)(pp + 16);
#pragma unroll
      for (int s = 0; s < 16; ++s)
        vb.v[s] = v[((size_t)bh * L + kbase + ks * 32 + ((s < 8) ? hl * 8 + s : 16 + hl * 8 + s - 8)) * HD + w * 16 + ln];
      o = wmma_bf16(pa.v, vb.v, o);
    }
  }

  // ---- Epilogue: normalize by running sum, add residual, scatter to x1 ----
  const int b  = bh / Hn;
  const int hh = bh % Hn;
  const int nn = w * 16 + ln;
#pragma unroll
  for (int r = 0; r < 8; ++r) {
    const int m = hl * 8 + r;
    const size_t idx = ((size_t)(b * L + l0 + m)) * E + hh * HD + nn;
    x1[idx] = o[r] / rs[m] + xres[idx];
  }
}

// ---------------------------------------------------------------------------
// Host launcher
// ---------------------------------------------------------------------------
extern "C" void kernel_launch(void* const* d_in, const int* /*in_sizes*/, int /*n_in*/,
                              void* d_out, int /*out_size*/, void* d_ws,
                              size_t /*ws_size*/, hipStream_t stream) {
  const float* x    = (const float*)d_in[0];
  const float* g    = (const float*)d_in[1];
  const float* wq   = (const float*)d_in[2];
  const float* wk   = (const float*)d_in[3];
  const float* wv   = (const float*)d_in[4];
  const float* w1   = (const float*)d_in[5];
  const float* b1   = (const float*)d_in[6];
  const float* wg   = (const float*)d_in[7];
  const float* bg   = (const float*)d_in[8];
  const float* wl   = (const float*)d_in[9];
  const float* bl   = (const float*)d_in[10];
  const float* beta = (const float*)d_in[11];
  const float* w2   = (const float*)d_in[12];
  const float* b2   = (const float*)d_in[13];

  constexpr int Bb = 2, L = 2048, E = 2048, Hn = 16, HD = 128, F = 4096;
  constexpr int M = Bb * L;  // 4096 rows

  char*  ws  = (char*)d_ws;
  size_t off = 0;
  auto take = [&](size_t bytes) -> void* {
    void* p = (void*)(ws + off);
    off += (bytes + 255) & ~(size_t)255;
    return p;
  };

  bf16_t* wqb  = (bf16_t*)take((size_t)E * E * 2);
  bf16_t* wkb  = (bf16_t*)take((size_t)E * E * 2);
  bf16_t* wvb  = (bf16_t*)take((size_t)E * E * 2);
  bf16_t* w1b  = (bf16_t*)take((size_t)E * F * 2);
  bf16_t* wgb  = (bf16_t*)take((size_t)F * F * 2);
  bf16_t* wlb  = (bf16_t*)take((size_t)F * F * 2);
  bf16_t* w2b  = (bf16_t*)take((size_t)F * E * 2);
  bf16_t* hbuf = (bf16_t*)take((size_t)M * E * 2);
  bf16_t* qbuf  = (bf16_t*)take((size_t)M * E * 2);  // contiguous with kTbuf
  bf16_t* kTbuf = (bf16_t*)take((size_t)M * E * 2);
  bf16_t* vbuf  = (bf16_t*)take((size_t)M * E * 2);
  bf16_t* gatebuf = (bf16_t*)take((size_t)M * F * 2);
  bf16_t* linbuf  = (bf16_t*)take((size_t)M * F * 2);
  float*  x1buf   = (float*)take((size_t)M * E * 4);
  float*  ropeTab = (float*)take((size_t)L * HD * 4);
  bf16_t* hhbuf = qbuf;  // reuse q+kT region (dead after attention) = M*F bf16
  (void)kTbuf;

  const dim3 blk(256);

  // 0) RoPE diagonal table (1 MB; L2-resident)
  rope_table_kernel<<<(L * HD + 255) / 256, blk, 0, stream>>>(ropeTab, L, HD);

  // 1) weights -> bf16 (126 MB total; resident in the 192 MB L2 afterwards)
  f32_to_bf16_kernel<<<2048, blk, 0, stream>>>(wq, wqb, (size_t)E * E);
  f32_to_bf16_kernel<<<2048, blk, 0, stream>>>(wk, wkb, (size_t)E * E);
  f32_to_bf16_kernel<<<2048, blk, 0, stream>>>(wv, wvb, (size_t)E * E);
  f32_to_bf16_kernel<<<2048, blk, 0, stream>>>(w1, w1b, (size_t)E * F);
  f32_to_bf16_kernel<<<4096, blk, 0, stream>>>(wg, wgb, (size_t)F * F);
  f32_to_bf16_kernel<<<4096, blk, 0, stream>>>(wl, wlb, (size_t)F * F);
  f32_to_bf16_kernel<<<2048, blk, 0, stream>>>(w2, w2b, (size_t)F * E);

  // 2) h = rmsnorm(x, g) -> bf16
  rmsnorm_bf16_kernel<<<M, blk, 0, stream>>>(x, g, hbuf, E);

  // 3) QKV projections with fused RoPE + per-head layout scatter
  const dim3 g1(E / 128, M / 128);
  gemm_bf16_kernel<EP_ROPE_Q><<<g1, blk, 0, stream>>>(
      M, E, E, hbuf, wqb, nullptr, nullptr, ropeTab, nullptr, qbuf, L, Hn, HD);
  gemm_bf16_kernel<EP_ROPE_KT><<<g1, blk, 0, stream>>>(
      M, E, E, hbuf, wkb, nullptr, nullptr, ropeTab, nullptr, kTbuf, L, Hn, HD);
  gemm_bf16_kernel<EP_VHEAD><<<g1, blk, 0, stream>>>(
      M, E, E, hbuf, wvb, nullptr, nullptr, nullptr, nullptr, vbuf, L, Hn, HD);

  // 4) causal attention + residual -> x1
  const dim3 ga(L / 16, Bb * Hn);
  attention_kernel<<<ga, blk, 0, stream>>>(qbuf, kTbuf, vbuf, x, x1buf, L, Hn, HD, E);

  // 5) h2 = rmsnorm(x1, g) -> bf16 (reuse hbuf)
  rmsnorm_bf16_kernel<<<M, blk, 0, stream>>>(x1buf, g, hbuf, E);

  // 6) FFN
  const dim3 g2(F / 128, M / 128);
  gemm_bf16_kernel<EP_BF16_BIAS><<<g2, blk, 0, stream>>>(
      M, F, E, hbuf, w1b, b1, nullptr, nullptr, nullptr, hhbuf, L, Hn, HD);
  gemm_bf16_kernel<EP_BF16_BIAS><<<g2, blk, 0, stream>>>(
      M, F, F, hhbuf, wgb, bg, nullptr, nullptr, nullptr, gatebuf, L, Hn, HD);
  gemm_bf16_kernel<EP_BF16_BIAS><<<g2, blk, 0, stream>>>(
      M, F, F, hhbuf, wlb, bl, nullptr, nullptr, nullptr, linbuf, L, Hn, HD);
  swiglu_kernel<<<4096, blk, 0, stream>>>(gatebuf, linbuf, beta, (size_t)M * F);

  // 7) out = act @ w2 + b2 + x1  (fp32 to d_out)
  const dim3 g3(E / 128, M / 128);
  gemm_bf16_kernel<EP_F32_BIAS_RES><<<g3, blk, 0, stream>>>(
      M, E, F, gatebuf, w2b, b2, x1buf, nullptr, (float*)d_out, nullptr, L, Hn, HD);
}
